// WeightGenerator_80994493268385
// MI455X (gfx1250) — matrix operand order repaired and checked by reading, not compiled, not run on hardware
//
#include <hip/hip_runtime.h>

#define CHUNK_    64
#define CODE_DIM_ 64
#define HIDDEN_   512
#define ENC_DIM_  13
#define IN_DIM_   154
#define K1PAD_    160   // IN_DIM padded to multiple of 32
#define MTILE_    64    // tokens per workgroup
#define PI_F 3.14159265358979323846f

typedef __attribute__((ext_vector_type(16))) __bf16          v16bf;
typedef __attribute__((ext_vector_type(8)))  float           v8f;
typedef __attribute__((ext_vector_type(16))) unsigned short  v16u;
typedef __attribute__((ext_vector_type(8)))  unsigned short  v8u;

union AFrag { v16u v; struct { v8u lo; v8u hi; } h; };

static __device__ __forceinline__ unsigned short f2bf(float f) {
  unsigned u = __float_as_uint(f);
  unsigned r = 0x7FFFu + ((u >> 16) & 1u);   // round-to-nearest-even
  return (unsigned short)((u + r) >> 16);
}

// Branch-free GELU (tanh form). Uses v_exp_f32 / v_rcp_f32 TRANS ops which
// co-execute with WMMA on CDNA5; no exec-mask divergence in the hot loop.
static __device__ __forceinline__ float gelu_fast(float x) {
  const float c0 = 0.7978845608028654f;   // sqrt(2/pi)
  const float c1 = 0.044715f;
  float y  = c0 * (x + c1 * x * x * x);
  float ay = __builtin_fabsf(y);
  float e  = __expf(-2.0f * ay);          // in (0,1], never overflows
  float t  = (1.0f - e) * __builtin_amdgcn_rcpf(1.0f + e);  // tanh(|y|)
  t = __builtin_copysignf(t, y);
  return 0.5f * x * (1.0f + t);
}

static __device__ __forceinline__ v16bf load_b_frag(const unsigned short* p) {
  return __builtin_bit_cast(v16bf, *(const v16u*)p);
}

// ---------------------------------------------------------------------------
// Prep: transpose f32 weights (K x N) -> bf16 (N x K), pad W1 K to 160.
// ---------------------------------------------------------------------------
__global__ void prep_weights(const float* __restrict__ W1, const float* __restrict__ W2,
                             const float* __restrict__ W3, const float* __restrict__ W4,
                             unsigned short* __restrict__ w1t, unsigned short* __restrict__ w2t,
                             unsigned short* __restrict__ w3t, unsigned short* __restrict__ w4t) {
  const int s1 = HIDDEN_ * K1PAD_;
  const int s2 = HIDDEN_ * HIDDEN_;
  const int s4 = CHUNK_ * HIDDEN_;
  int i = blockIdx.x * blockDim.x + threadIdx.x;
  if (i < s1) {
    int n = i / K1PAD_, k = i % K1PAD_;
    float v = (k < IN_DIM_) ? W1[k * HIDDEN_ + n] : 0.0f;
    w1t[n * K1PAD_ + k] = f2bf(v);
  } else if (i < s1 + s2) {
    int j = i - s1; int n = j / HIDDEN_, k = j % HIDDEN_;
    w2t[n * HIDDEN_ + k] = f2bf(W2[k * HIDDEN_ + n]);
  } else if (i < s1 + 2 * s2) {
    int j = i - s1 - s2; int n = j / HIDDEN_, k = j % HIDDEN_;
    w3t[n * HIDDEN_ + k] = f2bf(W3[k * HIDDEN_ + n]);
  } else if (i < s1 + 2 * s2 + s4) {
    int j = i - s1 - 2 * s2; int n = j / HIDDEN_, k = j % HIDDEN_;
    w4t[n * HIDDEN_ + k] = f2bf(W4[k * CHUNK_ + n]);
  }
}

// ---------------------------------------------------------------------------
// One hidden layer: H_out = gelu(A(64xK) @ Wt^T + b), bf16 tiles in LDS.
// Wave w owns columns [w*64, w*64+64): 4 N-tiles of 16; 4 M-tiles cover 64 rows.
// ---------------------------------------------------------------------------
template <int K, int LDS_K>
static __device__ __forceinline__ void
layer_lds(const unsigned short* __restrict__ Ain,
          const unsigned short* __restrict__ Wt,
          const float* __restrict__ bias,
          unsigned short* __restrict__ Hout,
          int wave, int ln, int hi) {
  for (int nt = 0; nt < 4; ++nt) {
    const int n0 = wave * 64 + nt * 16;
    v8f acc[4] = {};
    for (int k0 = 0; k0 < K; k0 += 32) {
      // B fragment: col (n0+ln), K = k0+hi*16 .. +15  (contiguous in transposed W)
      v16bf b = load_b_frag(Wt + (size_t)(n0 + ln) * K + k0 + hi * 16);
#pragma unroll
      for (int mt = 0; mt < 4; ++mt) {
        // A fragment: row (mt*16+ln); K = k0+hi*16 .. +15 contiguous
        const unsigned short* ap = Ain + (mt * 16 + ln) * LDS_K + k0 + hi * 16;
        AFrag af;
        af.h.lo = *(const v8u*)ap;
        af.h.hi = *(const v8u*)(ap + 8);
        v16bf a = __builtin_bit_cast(v16bf, af.v);
        acc[mt] = __builtin_amdgcn_wmma_f32_16x16x32_bf16(
            false, a, false, b, (short)0, acc[mt], false, false);
      }
    }
    const float bn = bias[n0 + ln];
#pragma unroll
    for (int mt = 0; mt < 4; ++mt)
#pragma unroll
      for (int r = 0; r < 8; ++r) {
        // C layout: m = r + hi*8 (per VGPR), n = lane&15
        float v = gelu_fast(acc[mt][r] + bn);
        Hout[(mt * 16 + r + hi * 8) * HIDDEN_ + n0 + ln] = f2bf(v);
      }
  }
}

// ---------------------------------------------------------------------------
// Fused MLP over one 64-token tile.
// ---------------------------------------------------------------------------
__global__ __launch_bounds__(256) void
mlp_fused(const float* __restrict__ layer_code, const float* __restrict__ type_embed,
          const float* __restrict__ b1, const float* __restrict__ b2,
          const float* __restrict__ b3, const float* __restrict__ b4,
          const unsigned short* __restrict__ w1t, const unsigned short* __restrict__ w2t,
          const unsigned short* __restrict__ w3t, const unsigned short* __restrict__ w4t,
          const float* __restrict__ out_scale_p, const int* __restrict__ type_idx_p,
          const int* __restrict__ rows_p, const int* __restrict__ cols_p,
          float* __restrict__ out) {
  __shared__ __attribute__((aligned(32))) unsigned short Abuf[MTILE_ * K1PAD_];
  __shared__ __attribute__((aligned(32))) unsigned short Hping[MTILE_ * HIDDEN_];
  __shared__ __attribute__((aligned(32))) unsigned short Hpong[MTILE_ * HIDDEN_];

  const int tid  = threadIdx.x;
  const int wave = tid >> 5;
  const int lane = tid & 31;
  const int ln   = lane & 15;
  const int hi   = lane >> 4;
  const int tile = blockIdx.x;

  const int rows = rows_p[0];
  const int cols = cols_p[0];
  const int n_chunks = cols / CHUNK_;
  const int tidx = type_idx_p[0];
  const float inv_r = 1.0f / (float)((rows - 1) > 1 ? (rows - 1) : 1);
  const float inv_c = 1.0f / (float)((cols - 1) > 1 ? (cols - 1) : 1);

  // ---- build 64x160 bf16 input tile in LDS ----
  for (int e = tid; e < MTILE_ * K1PAD_; e += 256) {
    const int m = e / K1PAD_, j = e % K1PAD_;
    const int t = tile * MTILE_ + m;
    const int row = t / n_chunks, ch = t % n_chunks;
    float v;
    if (j < CODE_DIM_) {
      v = layer_code[j];
    } else if (j < 2 * CODE_DIM_) {
      v = type_embed[tidx * CODE_DIM_ + (j - CODE_DIM_)];
    } else if (j < 2 * CODE_DIM_ + 2 * ENC_DIM_) {
      const int je = j - 2 * CODE_DIM_;
      const bool is_row = (je < ENC_DIM_);
      const float x = is_row ? ((float)row * inv_r)
                             : (((float)ch * CHUNK_ + (CHUNK_ / 2)) * inv_c);
      const int j13 = is_row ? je : je - ENC_DIM_;
      if (j13 == 0) {
        v = x;
      } else {
        const int q = (j13 - 1) >> 1;
        const float ang = PI_F * x * (float)(1 << q);
        v = ((j13 - 1) & 1) ? __cosf(ang) : __sinf(ang);
      }
    } else {
      v = 0.0f;
    }
    Abuf[e] = f2bf(v);
  }
  __syncthreads();

  layer_lds<K1PAD_, K1PAD_>(Abuf, w1t, b1, Hping, wave, ln, hi);
  __syncthreads();
  layer_lds<HIDDEN_, HIDDEN_>(Hping, w2t, b2, Hpong, wave, ln, hi);
  __syncthreads();
  layer_lds<HIDDEN_, HIDDEN_>(Hpong, w3t, b3, Hping, wave, ln, hi);
  __syncthreads();

  // ---- layer 4: N = 64 -> one N-tile per wave (waves 0..3) ----
  if (wave < 4) {
    const int n0 = wave * 16;
    v8f acc[4] = {};
    for (int k0 = 0; k0 < HIDDEN_; k0 += 32) {
      v16bf b = load_b_frag(w4t + (size_t)(n0 + ln) * HIDDEN_ + k0 + hi * 16);
#pragma unroll
      for (int mt = 0; mt < 4; ++mt) {
        const unsigned short* ap = Hping + (mt * 16 + ln) * HIDDEN_ + k0 + hi * 16;
        AFrag af;
        af.h.lo = *(const v8u*)ap;
        af.h.hi = *(const v8u*)(ap + 8);
        v16bf a = __builtin_bit_cast(v16bf, af.v);
        acc[mt] = __builtin_amdgcn_wmma_f32_16x16x32_bf16(
            false, a, false, b, (short)0, acc[mt], false, false);
      }
    }
    const float bn = b4[n0 + ln];
    const float sc = out_scale_p[0];
#pragma unroll
    for (int mt = 0; mt < 4; ++mt)
#pragma unroll
      for (int r = 0; r < 8; ++r) {
        const int m = mt * 16 + r + hi * 8;
        const size_t t = (size_t)tile * MTILE_ + m;
        out[t * CHUNK_ + n0 + ln] = sc * (acc[mt][r] + bn);
      }
  }
}

// ---------------------------------------------------------------------------
extern "C" void kernel_launch(void* const* d_in, const int* in_sizes, int n_in,
                              void* d_out, int out_size, void* d_ws, size_t ws_size,
                              hipStream_t stream) {
  const float* layer_code = (const float*)d_in[0];
  const float* type_embed = (const float*)d_in[1];
  const float* W1 = (const float*)d_in[2];
  const float* b1 = (const float*)d_in[3];
  const float* W2 = (const float*)d_in[4];
  const float* b2 = (const float*)d_in[5];
  const float* W3 = (const float*)d_in[6];
  const float* b3 = (const float*)d_in[7];
  const float* W4 = (const float*)d_in[8];
  const float* b4 = (const float*)d_in[9];
  const float* out_scale = (const float*)d_in[10];
  const int* type_idx = (const int*)d_in[11];
  const int* rows = (const int*)d_in[12];
  const int* cols = (const int*)d_in[13];
  float* out = (float*)d_out;

  unsigned short* w1t = (unsigned short*)d_ws;                 // 512*160
  unsigned short* w2t = w1t + HIDDEN_ * K1PAD_;                // 512*512
  unsigned short* w3t = w2t + HIDDEN_ * HIDDEN_;               // 512*512
  unsigned short* w4t = w3t + HIDDEN_ * HIDDEN_;               // 64*512

  const int prep_total = HIDDEN_ * K1PAD_ + 2 * HIDDEN_ * HIDDEN_ + CHUNK_ * HIDDEN_;
  prep_weights<<<(prep_total + 255) / 256, 256, 0, stream>>>(W1, W2, W3, W4,
                                                             w1t, w2t, w3t, w4t);

  const int tiles = out_size / (MTILE_ * CHUNK_);   // 64 tokens * 64 outputs per tile
  mlp_fused<<<tiles, 256, 0, stream>>>(layer_code, type_embed, b1, b2, b3, b4,
                                       w1t, w2t, w3t, w4t, out_scale, type_idx,
                                       rows, cols, out);
}
